// LIF_hh_neuron_67010079752677
// MI455X (gfx1250) — matrix operand order repaired and checked by reading, not compiled, not run on hardware
//
#include <hip/hip_runtime.h>

// ---------------------------------------------------------------------------
// Fused 3xGEMM + LIF spiking recurrence for MI455X (gfx1250, wave32, WMMA).
//
// Roofline: weights 201MB (read once) + spikes 252MB (written once) at
// 23.3 TB/s ~= 20us floor; 25.8 GFLOP of bf16 WMMA is far cheaper. So the
// kernel is structured to be pure-streaming: each W element is loaded from
// HBM exactly once (one workgroup owns all 256 batch rows for its 16
// channels), x (4MB) stays L2-resident, and the LIF recurrence is fused so
// GEMM results never leave registers.
// ---------------------------------------------------------------------------

typedef __attribute__((ext_vector_type(4)))  float  f32x4;
typedef __attribute__((ext_vector_type(8)))  float  v8f;
typedef __attribute__((ext_vector_type(4)))  __bf16 bf16x4;
typedef __attribute__((ext_vector_type(8)))  __bf16 bf16x8;
typedef __attribute__((ext_vector_type(16))) __bf16 v16bf;

#define LIF_THRESH 0.8f
#define LIF_DECAY  0.2f

constexpr int BATCH = 256;
constexpr int KDIM  = 4096;
constexpr int COUT  = 4096;
constexpr int NT    = 16;   // channels per workgroup
constexpr int LDA   = 40;   // padded LDS row stride in bf16 elems (80B, 16B-aligned rows)

union FragBF16 { v16bf v; bf16x8 h[2]; };

__global__ __launch_bounds__(256, 1)
void lif3fc_fused_kernel(const float* __restrict__ x,
                         const float* __restrict__ W1, const float* __restrict__ b1,
                         const float* __restrict__ W2, const float* __restrict__ b2,
                         const float* __restrict__ W3, const float* __restrict__ b3,
                         const float* __restrict__ Wlif, const float* __restrict__ blif,
                         const int*   __restrict__ winsp,
                         float* __restrict__ out)
{
    __shared__ __align__(16) __bf16 Asm[BATCH * LDA];      // 256 x 32 bf16 tile (padded)
    __shared__ __align__(16) __bf16 Bsm[3 * NT * LDA];     // 3 x 16 x 32 bf16 tiles

    const int tid    = threadIdx.x;
    const int lane   = tid & 31;
    const int wave   = tid >> 5;          // 8 waves; wave w owns batch rows [32w, 32w+32)
    const int laneHi = lane >> 4;         // 0: lanes 0-15, 1: lanes 16-31
    const int l15    = lane & 15;
    const int cBase  = blockIdx.x * NT;   // this WG's 16 output channels

    // --- accumulators: [m-subtile (0..1)][matrix (0..2)], 16x16 f32 C/D each ---
    v8f acc[2][3];
    #pragma unroll
    for (int s = 0; s < 2; ++s)
        #pragma unroll
        for (int j = 0; j < 3; ++j)
            #pragma unroll
            for (int r = 0; r < 8; ++r)
                acc[s][j][r] = 0.0f;

    // --- per-lane LDS fragment pointers (fixed across the K loop) ---
    // A 16x32 bf16 layout: lanes 0-15 -> M=lane, K in {0..7,16..23};
    //                      lanes 16-31 -> M=lane-16, K in {8..15,24..31}.
    const int rowA = wave * 32 + l15;
    const int aoff = laneHi ? 8 : 0;
    const bf16x8* aLo0 = (const bf16x8*)&Asm[(rowA     ) * LDA + aoff];
    const bf16x8* aHi0 = (const bf16x8*)&Asm[(rowA     ) * LDA + aoff + 16];
    const bf16x8* aLo1 = (const bf16x8*)&Asm[(rowA + 16) * LDA + aoff];
    const bf16x8* aHi1 = (const bf16x8*)&Asm[(rowA + 16) * LDA + aoff + 16];
    // B 32x16 bf16 layout: lane n<16 -> N=n, K in {0..7,16..23} ; lanes 16-31 -> N=n-16, K in {8..15,24..31}.
    const int boff = laneHi ? 8 : 0;
    const bf16x8* bLo[3];
    const bf16x8* bHi[3];
    #pragma unroll
    for (int j = 0; j < 3; ++j) {
        bLo[j] = (const bf16x8*)&Bsm[(j * NT + l15) * LDA + boff];
        bHi[j] = (const bf16x8*)&Bsm[(j * NT + l15) * LDA + boff + 16];
    }

    // ================= K loop: 128 steps of K=32 =================
    for (int kb = 0; kb < KDIM; kb += 32) {
        __syncthreads();   // previous iteration's fragment reads done

        // ---- stage A tile: 256 rows x 32 k, fp32 -> bf16 ----
        {
            const int kc = tid & 7;        // 4-float chunk within the 32-wide k slab
            const int r0 = tid >> 3;       // 32 rows per pass, coalesced 128B per row
            #pragma unroll
            for (int i = 0; i < 8; ++i) {
                const int row = i * 32 + r0;
                f32x4 v = *(const f32x4*)&x[(size_t)row * KDIM + kb + kc * 4];
                *(bf16x4*)&Asm[row * LDA + kc * 4] = __builtin_convertvector(v, bf16x4);
            }
        }
        // ---- stage B tiles: 3 matrices x 16 ch x 32 k = 384 float4 chunks ----
        {
            int q = tid;
            #pragma unroll
            for (int it = 0; it < 2; ++it) {
                if (q < 384) {
                    const int j  = q >> 7;
                    const int rm = q & 127;
                    const int ch = rm >> 3;
                    const int kc = rm & 7;
                    const float* Wj = (j == 0) ? W1 : ((j == 1) ? W2 : W3);
                    const float* gp = &Wj[(size_t)(cBase + ch) * KDIM + kb + kc * 4];
                    f32x4 v = *(const f32x4*)gp;
                    *(bf16x4*)&Bsm[(j * NT + ch) * LDA + kc * 4] =
                        __builtin_convertvector(v, bf16x4);
                    if (kb + 32 < KDIM)
                        __builtin_prefetch(gp + 32, 0, 1);   // next K-slab of the HBM stream
                }
                q += 256;
            }
        }
        __syncthreads();

        // ---- fragments + 6 WMMAs (A frags shared across the 3 matrices) ----
        FragBF16 fa0, fa1;
        fa0.h[0] = *aLo0; fa0.h[1] = *aHi0;
        fa1.h[0] = *aLo1; fa1.h[1] = *aHi1;
        #pragma unroll
        for (int j = 0; j < 3; ++j) {
            FragBF16 fb;
            fb.h[0] = *bLo[j]; fb.h[1] = *bHi[j];
            acc[0][j] = __builtin_amdgcn_wmma_f32_16x16x32_bf16(
                false, fa0.v, false, fb.v, (short)0, acc[0][j], false, false);
            acc[1][j] = __builtin_amdgcn_wmma_f32_16x16x32_bf16(
                false, fa1.v, false, fb.v, (short)0, acc[1][j], false, false);
        }
    }

    // ================= bias + fused LIF recurrence (all in registers) ========
    // C/D layout: lane (l15) -> channel column; VGPR r -> batch row r + 8*laneHi.
    const int   c    = cBase + l15;
    const float b1c  = b1[c], b2c = b2[c], b3c = b3[c];
    const float wl0  = Wlif[0], wl1 = Wlif[1], wl2 = Wlif[2];
    const float bl   = blif[0];
    const int   wins = *winsp;

    #pragma unroll
    for (int s = 0; s < 2; ++s) {
        #pragma unroll
        for (int r = 0; r < 8; ++r) {
            const float y0 = acc[s][0][r] + b1c;
            const float y1 = acc[s][1][r] + b2c;
            const float y2 = acc[s][2][r] + b3c;
            const int   brow = wave * 32 + s * 16 + laneHi * 8 + r;
            float* op = out + ((size_t)brow * wins) * (size_t)(COUT * 4) + (size_t)c * 4;

            float m0 = 0.f, m1 = 0.f, m2 = 0.f, m3 = 0.f;
            for (int t = 0; t < wins; ++t) {
                // spike from previous mem (recomputed; t=0: mem=0 -> spike=0)
                const float s0 = (m0 > LIF_THRESH) ? 1.f : 0.f;
                const float s1 = (m1 > LIF_THRESH) ? 1.f : 0.f;
                const float s2 = (m2 > LIF_THRESH) ? 1.f : 0.f;
                const float s3 = (m3 > LIF_THRESH) ? 1.f : 0.f;
                const float inner = m0 * wl0 + m1 * wl1 + m2 * wl2 + bl;
                m0 = m0 * LIF_DECAY * (1.f - s0) + y0;
                m1 = m1 * LIF_DECAY * (1.f - s1) + y1;
                m2 = m2 * LIF_DECAY * (1.f - s2) + y2;
                m3 = m3 * LIF_DECAY * (1.f - s3) + inner;
                f32x4 sp;
                sp.x = (m0 > LIF_THRESH) ? 1.f : 0.f;
                sp.y = (m1 > LIF_THRESH) ? 1.f : 0.f;
                sp.z = (m2 > LIF_THRESH) ? 1.f : 0.f;
                sp.w = (m3 > LIF_THRESH) ? 1.f : 0.f;
                *(f32x4*)op = sp;                 // coalesced 16B per lane, 256B bursts
                op += COUT * 4;                   // next timestep
            }
        }
    }
}

extern "C" void kernel_launch(void* const* d_in, const int* in_sizes, int n_in,
                              void* d_out, int out_size, void* d_ws, size_t ws_size,
                              hipStream_t stream) {
    const float* x    = (const float*)d_in[0];
    const float* W1   = (const float*)d_in[1];
    const float* b1   = (const float*)d_in[2];
    const float* W2   = (const float*)d_in[3];
    const float* b2   = (const float*)d_in[4];
    const float* W3   = (const float*)d_in[5];
    const float* b3   = (const float*)d_in[6];
    const float* Wlif = (const float*)d_in[7];
    const float* blif = (const float*)d_in[8];
    const int*   wins = (const int*)d_in[9];
    float* out = (float*)d_out;

    dim3 grid(COUT / NT);   // 256 workgroups, each owns 16 channels x all 256 rows
    dim3 block(256);        // 8 wave32s
    lif3fc_fused_kernel<<<grid, block, 0, stream>>>(
        x, W1, b1, W2, b2, W3, b3, Wlif, blif, wins, out);
}
